// GraphPerformer_61160334295742
// MI455X (gfx1250) — matrix operand order repaired and checked by reading, not compiled, not run on hardware
//
#include <hip/hip_runtime.h>
#include <hip/hip_bf16.h>

// ---------------------------------------------------------------------------
// GraphPerformer on gfx1250 (MI455X).
#define N_NODES 2048
#define N_TOKENS 32
#define DIM 256
#define HEADS 8
#define DH 32
#define INNER 256
#define NB 64
#define NLAYER 4
#define FF 1024
#define N_EDGES 16384
#define ROWS (N_NODES * N_TOKENS)   // 65536

typedef __attribute__((ext_vector_type(16))) __bf16 v16bf;
typedef __attribute__((ext_vector_type(8)))  float  v8f;

union FragBF {
    v16bf v;
    uint4 q[2];
    unsigned short s[16];
};

// ---- helpers --------------------------------------------------------------
static __device__ __forceinline__ unsigned short f2bf(float f) {
    unsigned int u = __float_as_uint(f);
    u += 0x7FFFu + ((u >> 16) & 1u);   // round-to-nearest-even
    return (unsigned short)(u >> 16);
}
static __device__ __forceinline__ float bf2f(unsigned short h) {
    return __uint_as_float(((unsigned int)h) << 16);
}
static __device__ __forceinline__ float gelu_exact(float x) {
    return 0.5f * x * (1.0f + erff(x * 0.70710678118654752440f));
}
// order-preserving float->uint for atomicMax
static __device__ __forceinline__ unsigned int fenc(float f) {
    unsigned int u = __float_as_uint(f);
    return u ^ ((unsigned int)(((int)u) >> 31) | 0x80000000u);
}
static __device__ __forceinline__ float fdec(unsigned int e) {
    unsigned int u = (e & 0x80000000u) ? (e ^ 0x80000000u) : ~e;
    return __uint_as_float(u);
}

// ---- generic zero ---------------------------------------------------------
__global__ void zero_kernel(float* __restrict__ p, long long n) {
    long long i = (long long)blockIdx.x * 256 + threadIdx.x;
    if (i < n) p[i] = 0.0f;
}

// ---- degree ---------------------------------------------------------------
__global__ void deg_kernel(const long long* __restrict__ ei, float* __restrict__ deg) {
    int e = blockIdx.x * 256 + threadIdx.x;
    if (e < N_EDGES) {
        int dst = (int)ei[N_EDGES + e];
        atomicAdd(&deg[dst], 1.0f);
    }
}

// ---- pack fp32 weight (K x N row-major) into WMMA B fragment layout -------
__global__ __launch_bounds__(256) void pack_b_kernel(const float* __restrict__ W,
                                                     unsigned short* __restrict__ out,
                                                     int K, int N) {
    long long i = (long long)blockIdx.x * 256 + threadIdx.x;
    long long total = (long long)K * N;
    if (i >= total) return;
    int e      = (int)(i & 15);
    int lane   = (int)((i >> 4) & 31);
    long long tile = i >> 9;                 // kt*nTiles + nt
    int nTiles = N >> 4;
    int kt = (int)(tile / nTiles);
    int nt = (int)(tile % nTiles);
    int halfB = lane >> 4, c = lane & 15;
    int k = kt * 32 + halfB * 8 + (e & 7) + (e >> 3) * 16;
    int n = nt * 16 + c;
    out[i] = f2bf(W[(long long)k * N + n]);
}

// ---- LayerNorm (fp32 in) -> bf16 out, one wave32 per 256-wide row ---------
__global__ __launch_bounds__(256) void ln_bf16_kernel(const float* __restrict__ x,
                                                      const float* __restrict__ g,
                                                      const float* __restrict__ b,
                                                      unsigned short* __restrict__ out) {
    int wave = threadIdx.x >> 5;
    int lane = threadIdx.x & 31;
    long long row = (long long)blockIdx.x * 8 + wave;
    const float* xr = x + row * DIM + lane * 8;
    float v[8];
    float s = 0.0f, s2 = 0.0f;
    #pragma unroll
    for (int i = 0; i < 8; ++i) { v[i] = xr[i]; s += v[i]; s2 += v[i] * v[i]; }
    #pragma unroll
    for (int m = 16; m >= 1; m >>= 1) {
        s  += __shfl_xor(s,  m, 32);
        s2 += __shfl_xor(s2, m, 32);
    }
    float mean = s * (1.0f / DIM);
    float var  = s2 * (1.0f / DIM) - mean * mean;
    float rstd = rsqrtf(var + 1e-5f);
    unsigned short o[8];
    #pragma unroll
    for (int i = 0; i < 8; ++i) {
        int c = lane * 8 + i;
        o[i] = f2bf((v[i] - mean) * rstd * g[c] + b[c]);
    }
    *reinterpret_cast<uint4*>(out + row * DIM + lane * 8) = *reinterpret_cast<const uint4*>(o);
}

// ---- bf16 WMMA GEMM, 64x64 tile per wave (4Mx4N), double-buffered over K --
// Block = 8 waves arranged 4M x 2N -> block tile 256 rows x 128 cols.
// grid = (M/256, N/128). Requires K % 64 == 0 (K = 256 or 1024 here).
// mode 0: out_bf = bf16(val)   mode 1: out_bf = bf16(gelu(val))
// mode 2: out_f  = resid + val (fp32 residual add)
// deg != null: bias scaled by [deg[node]>0], node = row/32
__global__ __launch_bounds__(256) void gemm_bf16_kernel(
        const unsigned short* __restrict__ A, const unsigned short* __restrict__ Bp,
        const float* __restrict__ bias, const float* __restrict__ deg,
        const float* __restrict__ resid, unsigned short* __restrict__ out_bf,
        float* __restrict__ out_f, int M, int N, int K, int mode) {
    const int lane = threadIdx.x & 31;
    const int wave = threadIdx.x >> 5;
    const int mBase = blockIdx.x * 256 + (wave >> 1) * 64;   // 4 M-subtiles of 16
    const int nBase = blockIdx.y * 128 + (wave & 1) * 64;    // 4 N-subtiles of 16
    const int nTiles = N >> 4;
    const int half = lane >> 4;
    const int r    = lane & 15;

    long long rowA[4];
    #pragma unroll
    for (int mt = 0; mt < 4; ++mt) rowA[mt] = (long long)(mBase + mt * 16 + r) * K;

    v8f acc[4][4];
    #pragma unroll
    for (int mt = 0; mt < 4; ++mt)
        #pragma unroll
        for (int nt = 0; nt < 4; ++nt)
            #pragma unroll
            for (int i = 0; i < 8; ++i) acc[mt][nt][i] = 0.0f;

    FragBF fa0[4], fb0[4], fa1[4], fb1[4];

    auto loadA = [&](FragBF* fa, int k0) {
        #pragma unroll
        for (int mt = 0; mt < 4; ++mt) {
            const unsigned short* p = A + rowA[mt] + k0 + half * 8;
            fa[mt].q[0] = *reinterpret_cast<const uint4*>(p);
            fa[mt].q[1] = *reinterpret_cast<const uint4*>(p + 16);
        }
    };
    auto loadB = [&](FragBF* fb, int k0) {
        const unsigned short* base =
            Bp + ((long long)((k0 >> 5) * nTiles + (nBase >> 4)) * 32 + lane) * 16;
        #pragma unroll
        for (int nt = 0; nt < 4; ++nt) {
            const uint4* pb = reinterpret_cast<const uint4*>(base + (long long)nt * 512);
            fb[nt].q[0] = pb[0];
            fb[nt].q[1] = pb[1];
        }
    };
    auto mma = [&](FragBF* fA, FragBF* fB) {
        #pragma unroll
        for (int mt = 0; mt < 4; ++mt)
            #pragma unroll
            for (int nt = 0; nt < 4; ++nt)
                acc[mt][nt] = __builtin_amdgcn_wmma_f32_16x16x32_bf16(
                    false, fA[mt].v, false, fB[nt].v, (short)0, acc[mt][nt], false, false);
    };

    loadA(fa0, 0);
    loadB(fb0, 0);
    for (int k0 = 0; k0 < K; k0 += 64) {
        loadA(fa1, k0 + 32);
        loadB(fb1, k0 + 32);
        if (k0 + 64 < K) __builtin_prefetch(A + rowA[0] + k0 + 64, 0, 3);
        mma(fa0, fb0);
        if (k0 + 64 < K) {
            loadA(fa0, k0 + 64);
            loadB(fb0, k0 + 64);
        }
        mma(fa1, fb1);
    }

    const int c = lane & 15;
    const int rbase = (lane >> 4) * 8;
    #pragma unroll
    for (int mt = 0; mt < 4; ++mt) {
        #pragma unroll
        for (int nt = 0; nt < 4; ++nt) {
            int n = nBase + nt * 16 + c;
            float bn = bias ? bias[n] : 0.0f;
            #pragma unroll
            for (int rr = 0; rr < 8; ++rr) {
                int m = mBase + mt * 16 + rbase + rr;
                float bscale = 1.0f;
                if (deg) bscale = (deg[m >> 5] > 0.0f) ? 1.0f : 0.0f;
                float val = acc[mt][nt][rr] + bn * bscale;
                long long idx = (long long)m * N + n;
                if (mode == 0)      out_bf[idx] = f2bf(val);
                else if (mode == 1) out_bf[idx] = f2bf(gelu_exact(val));
                else                out_f[idx]  = resid[idx] + val;
            }
        }
    }
}

// ---- graph mean-aggregation of h (bf16 -> fp32 atomics) -------------------
__global__ __launch_bounds__(256) void agg_kernel(const unsigned short* __restrict__ h,
                                                  const long long* __restrict__ ei,
                                                  float* __restrict__ hagg) {
    int e = blockIdx.x;
    int src = (int)ei[e];
    int dst = (int)ei[N_EDGES + e];
    const unsigned short* hs = h + (long long)src * (N_TOKENS * DIM);
    float* hd = hagg + (long long)dst * (N_TOKENS * DIM);
    for (int i = threadIdx.x; i < N_TOKENS * DIM; i += 256)
        atomicAdd(&hd[i], bf2f(hs[i]));
}

__global__ __launch_bounds__(256) void divdeg_kernel(const float* __restrict__ hagg,
                                                     const float* __restrict__ deg,
                                                     unsigned short* __restrict__ hg) {
    long long i = (long long)blockIdx.x * 256 + threadIdx.x;
    if (i >= (long long)ROWS * DIM) return;
    float d = fmaxf(deg[i >> 13], 1.0f);   // 32*256 = 8192 elems per node
    hg[i] = f2bf(hagg[i] / d);
}

// ---- FAVOR+ softmax kernel, query path (per-row max) ----------------------
__global__ __launch_bounds__(64) void favor_q_kernel(const unsigned short* __restrict__ qb,
                                                     const float* __restrict__ proj,
                                                     float* __restrict__ qh) {
    int b = blockIdx.x, hd = blockIdx.y;
    __shared__ float qs[N_TOKENS][DH];
    __shared__ float ps[NB][DH];
    __shared__ float dds[N_TOKENS][NB];
    __shared__ float diag[N_TOKENS];
    __shared__ float rowmax[N_TOKENS][2];
    int tid = threadIdx.x;
    const float dn = 0.42044820762685725f;  // 32^-0.25
    for (int i = tid; i < N_TOKENS * DH; i += 64)
        qs[i >> 5][i & 31] = bf2f(qb[((long long)(b * N_TOKENS + (i >> 5))) * INNER + hd * DH + (i & 31)]);
    for (int i = tid; i < NB * DH; i += 64) ps[i >> 5][i & 31] = proj[i];
    __syncthreads();
    if (tid < N_TOKENS) {
        float s = 0.0f;
        for (int d = 0; d < DH; ++d) { float t = qs[tid][d]; s += t * t; }
        diag[tid] = 0.5f * dn * dn * s;
    }
    __syncthreads();
    int n = tid >> 1, j0 = (tid & 1) * 32;
    float mx = -3.0e38f;
    for (int j = 0; j < 32; ++j) {
        float s = 0.0f;
        for (int d = 0; d < DH; ++d) s += qs[n][d] * dn * ps[j0 + j][d];
        dds[n][j0 + j] = s;
        mx = fmaxf(mx, s);
    }
    rowmax[n][tid & 1] = mx;
    __syncthreads();
    float m = fmaxf(rowmax[n][0], rowmax[n][1]);
    const float ratio = 0.125f, eps = 1e-4f;
    float* outp = qh + (((long long)(b * HEADS + hd)) * N_TOKENS + n) * NB + j0;
    for (int j = 0; j < 32; ++j)
        outp[j] = ratio * (expf(dds[n][j0 + j] - diag[n] - m) + eps);
}

// ---- FAVOR+ key path pass 1: dd + diag + global max -----------------------
__global__ __launch_bounds__(64) void favor_k1_kernel(const unsigned short* __restrict__ kb,
                                                      const float* __restrict__ proj,
                                                      float* __restrict__ kdd,
                                                      float* __restrict__ diagK,
                                                      unsigned int* __restrict__ gmax) {
    int b = blockIdx.x, hd = blockIdx.y;
    __shared__ float qs[N_TOKENS][DH];
    __shared__ float ps[NB][DH];
    __shared__ float red[64];
    int tid = threadIdx.x;
    const float dn = 0.42044820762685725f;
    for (int i = tid; i < N_TOKENS * DH; i += 64)
        qs[i >> 5][i & 31] = bf2f(kb[((long long)(b * N_TOKENS + (i >> 5))) * INNER + hd * DH + (i & 31)]);
    for (int i = tid; i < NB * DH; i += 64) ps[i >> 5][i & 31] = proj[i];
    __syncthreads();
    if (tid < N_TOKENS) {
        float s = 0.0f;
        for (int d = 0; d < DH; ++d) { float t = qs[tid][d]; s += t * t; }
        diagK[((long long)(b * HEADS + hd)) * N_TOKENS + tid] = 0.5f * dn * dn * s;
    }
    int n = tid >> 1, j0 = (tid & 1) * 32;
    float mx = -3.0e38f;
    float* outp = kdd + (((long long)(b * HEADS + hd)) * N_TOKENS + n) * NB + j0;
    for (int j = 0; j < 32; ++j) {
        float s = 0.0f;
        for (int d = 0; d < DH; ++d) s += qs[n][d] * dn * ps[j0 + j][d];
        outp[j] = s;
        mx = fmaxf(mx, s);
    }
    red[tid] = mx;
    __syncthreads();
    for (int off = 32; off >= 1; off >>= 1) {
        if (tid < off) red[tid] = fmaxf(red[tid], red[tid + off]);
        __syncthreads();
    }
    if (tid == 0) atomicMax(gmax, fenc(red[0]));
}

// ---- FAVOR+ key path pass 2: elementwise exp ------------------------------
__global__ __launch_bounds__(256) void favor_k2_kernel(float* __restrict__ kdd,
                                                       const float* __restrict__ diagK,
                                                       const unsigned int* __restrict__ gmax) {
    long long i = (long long)blockIdx.x * 256 + threadIdx.x;
    const long long total = (long long)N_NODES * HEADS * N_TOKENS * NB;
    if (i >= total) return;
    float m = fdec(*gmax);
    kdd[i] = 0.125f * (expf(kdd[i] - diagK[i >> 6] - m) + 1e-4f);
}

// ---- linear attention, one node per block ---------------------------------
__global__ __launch_bounds__(256) void linattn_kernel(const float* __restrict__ qh,
                                                      const float* __restrict__ kh,
                                                      const unsigned short* __restrict__ vb,
                                                      unsigned short* __restrict__ o) {
    int b = blockIdx.x;
    __shared__ float qs[N_TOKENS][NB];
    __shared__ float ks[N_TOKENS][NB];
    __shared__ float vs[N_TOKENS][DH];
    __shared__ float ctx[NB][DH];
    __shared__ float kcum[NB];
    __shared__ float dinv[N_TOKENS];
    for (int hd = 0; hd < HEADS; ++hd) {
        const float* qp = qh + ((long long)(b * HEADS + hd)) * (N_TOKENS * NB);
        const float* kp = kh + ((long long)(b * HEADS + hd)) * (N_TOKENS * NB);
        for (int i = threadIdx.x; i < N_TOKENS * NB; i += 256) {
            qs[i >> 6][i & 63] = qp[i];
            ks[i >> 6][i & 63] = kp[i];
        }
        for (int i = threadIdx.x; i < N_TOKENS * DH; i += 256)
            vs[i >> 5][i & 31] = bf2f(vb[((long long)(b * N_TOKENS + (i >> 5))) * INNER + hd * DH + (i & 31)]);
        __syncthreads();
        if (threadIdx.x < NB) {
            float s = 0.0f;
            for (int n = 0; n < N_TOKENS; ++n) s += ks[n][threadIdx.x];
            kcum[threadIdx.x] = s;
        }
        __syncthreads();
        if (threadIdx.x < N_TOKENS) {
            float s = 0.0f;
            for (int j = 0; j < NB; ++j) s += qs[threadIdx.x][j] * kcum[j];
            dinv[threadIdx.x] = 1.0f / s;
        }
        for (int i = threadIdx.x; i < NB * DH; i += 256) {
            int j = i >> 5, e = i & 31;
            float s = 0.0f;
            for (int n = 0; n < N_TOKENS; ++n) s += ks[n][j] * vs[n][e];
            ctx[j][e] = s;
        }
        __syncthreads();
        for (int i = threadIdx.x; i < N_TOKENS * DH; i += 256) {
            int n = i >> 5, e = i & 31;
            float s = 0.0f;
            for (int j = 0; j < NB; ++j) s += ctx[j][e] * qs[n][j];
            o[((long long)(b * N_TOKENS + n)) * INNER + hd * DH + e] = f2bf(s * dinv[n]);
        }
        __syncthreads();
    }
}

// ---------------------------------------------------------------------------
extern "C" void kernel_launch(void* const* d_in, const int* in_sizes, int n_in,
                              void* d_out, int out_size, void* d_ws, size_t ws_size,
                              hipStream_t stream) {
    (void)in_sizes; (void)n_in; (void)out_size; (void)ws_size;

    const float*     x_in  = (const float*)d_in[0];
    const long long* ei    = (const long long*)d_in[1];
    const float*     ln_ag = (const float*)d_in[2];
    const float*     ln_ab = (const float*)d_in[3];
    const float*     Wq    = (const float*)d_in[4];
    const float*     bq    = (const float*)d_in[5];
    const float*     Wk    = (const float*)d_in[6];
    const float*     bk    = (const float*)d_in[7];
    const float*     Wv    = (const float*)d_in[8];
    const float*     bv    = (const float*)d_in[9];
    const float*     Wo    = (const float*)d_in[10];
    const float*     bo    = (const float*)d_in[11];
    const float*     proj  = (const float*)d_in[12];
    const float*     ln_fg = (const float*)d_in[13];
    const float*     ln_fb = (const float*)d_in[14];
    const float*     W1    = (const float*)d_in[15];
    const float*     b1    = (const float*)d_in[16];
    const float*     W2    = (const float*)d_in[17];
    const float*     b2    = (const float*)d_in[18];

    float* xf = (float*)d_out;  // resident fp32 activation buffer == output

    // ---- workspace carve ----
    char* wp = (char*)d_ws;
    auto carve = [&](size_t bytes) -> void* {
        void* p = (void*)wp;
        wp += (bytes + 255) & ~(size_t)255;
        return p;
    };
    unsigned short* hbuf  = (unsigned short*)carve((size_t)ROWS * DIM * 2);
    unsigned short* hg    = (unsigned short*)carve((size_t)ROWS * DIM * 2);
    unsigned short* qb    = (unsigned short*)carve((size_t)ROWS * INNER * 2);
    unsigned short* kb    = (unsigned short*)carve((size_t)ROWS * INNER * 2);
    unsigned short* vb    = (unsigned short*)carve((size_t)ROWS * INNER * 2);
    unsigned short* obuf  = (unsigned short*)carve((size_t)ROWS * INNER * 2);
    unsigned short* ubuf  = (unsigned short*)carve((size_t)ROWS * FF * 2);
    float*          hagg  = (float*)carve((size_t)ROWS * DIM * 4);
    float*          qh    = (float*)carve((size_t)N_NODES * HEADS * N_TOKENS * NB * 4);
    float*          kh    = (float*)carve((size_t)N_NODES * HEADS * N_TOKENS * NB * 4);
    float*          diagK = (float*)carve((size_t)N_NODES * HEADS * N_TOKENS * 4);
    unsigned short* wq_p  = (unsigned short*)carve((size_t)NLAYER * DIM * INNER * 2);
    unsigned short* wk_p  = (unsigned short*)carve((size_t)NLAYER * DIM * INNER * 2);
    unsigned short* wv_p  = (unsigned short*)carve((size_t)NLAYER * DIM * INNER * 2);
    unsigned short* wo_p  = (unsigned short*)carve((size_t)NLAYER * INNER * DIM * 2);
    unsigned short* w1_p  = (unsigned short*)carve((size_t)NLAYER * DIM * FF * 2);
    unsigned short* w2_p  = (unsigned short*)carve((size_t)NLAYER * FF * DIM * 2);
    float*          deg   = (float*)carve((size_t)N_NODES * 4);
    unsigned int*   gmax  = (unsigned int*)carve(256);

    // ---- x -> working buffer ----
    hipMemcpyAsync(xf, x_in, (size_t)ROWS * DIM * 4, hipMemcpyDeviceToDevice, stream);

    // ---- pack weights to WMMA B layout (bf16) ----
    for (int l = 0; l < NLAYER; ++l) {
        pack_b_kernel<<<256, 256, 0, stream>>>(Wq + (size_t)l * DIM * INNER, wq_p + (size_t)l * DIM * INNER, DIM, INNER);
        pack_b_kernel<<<256, 256, 0, stream>>>(Wk + (size_t)l * DIM * INNER, wk_p + (size_t)l * DIM * INNER, DIM, INNER);
        pack_b_kernel<<<256, 256, 0, stream>>>(Wv + (size_t)l * DIM * INNER, wv_p + (size_t)l * DIM * INNER, DIM, INNER);
        pack_b_kernel<<<256, 256, 0, stream>>>(Wo + (size_t)l * INNER * DIM, wo_p + (size_t)l * INNER * DIM, INNER, DIM);
        pack_b_kernel<<<1024, 256, 0, stream>>>(W1 + (size_t)l * DIM * FF, w1_p + (size_t)l * DIM * FF, DIM, FF);
        pack_b_kernel<<<1024, 256, 0, stream>>>(W2 + (size_t)l * FF * DIM, w2_p + (size_t)l * FF * DIM, FF, DIM);
    }

    // ---- degree ----
    zero_kernel<<<(N_NODES + 255) / 256, 256, 0, stream>>>(deg, N_NODES);
    deg_kernel<<<N_EDGES / 256, 256, 0, stream>>>(ei, deg);

    auto gemm = [&](const unsigned short* A, const unsigned short* Bp, const float* bias,
                    const float* degp, const float* resid, unsigned short* obf, float* of,
                    int N, int K, int mode) {
        dim3 g(ROWS / 256, N / 128);
        gemm_bf16_kernel<<<g, 256, 0, stream>>>(A, Bp, bias, degp, resid, obf, of,
                                                ROWS, N, K, mode);
    };

    for (int l = 0; l < NLAYER; ++l) {
        // --- attention block ---
        ln_bf16_kernel<<<ROWS / 8, 256, 0, stream>>>(xf, ln_ag + l * DIM, ln_ab + l * DIM, hbuf);
        gemm(hbuf, wq_p + (size_t)l * DIM * INNER, bq + l * INNER, nullptr, nullptr, qb, nullptr, INNER, DIM, 0);
        if ((l & 1) == 0) {
            gemm(hbuf, wk_p + (size_t)l * DIM * INNER, bk + l * INNER, nullptr, nullptr, kb, nullptr, INNER, DIM, 0);
            gemm(hbuf, wv_p + (size_t)l * DIM * INNER, bv + l * INNER, nullptr, nullptr, vb, nullptr, INNER, DIM, 0);
        } else {
            // mean-aggregate h over incoming edges, then project (linearity of segment_sum)
            zero_kernel<<<(ROWS * DIM) / 256, 256, 0, stream>>>(hagg, (long long)ROWS * DIM);
            agg_kernel<<<N_EDGES, 256, 0, stream>>>(hbuf, ei, hagg);
            divdeg_kernel<<<(ROWS * DIM) / 256, 256, 0, stream>>>(hagg, deg, hg);
            gemm(hg, wk_p + (size_t)l * DIM * INNER, bk + l * INNER, deg, nullptr, kb, nullptr, INNER, DIM, 0);
            gemm(hg, wv_p + (size_t)l * DIM * INNER, bv + l * INNER, deg, nullptr, vb, nullptr, INNER, DIM, 0);
        }
        favor_q_kernel<<<dim3(N_NODES, HEADS), 64, 0, stream>>>(qb, proj + (size_t)l * NB * DH, qh);
        zero_kernel<<<1, 256, 0, stream>>>((float*)gmax, 1);   // 0x0 == minimal encoded key
        favor_k1_kernel<<<dim3(N_NODES, HEADS), 64, 0, stream>>>(kb, proj + (size_t)l * NB * DH, kh, diagK, gmax);
        favor_k2_kernel<<<(N_NODES * HEADS * N_TOKENS * NB) / 256, 256, 0, stream>>>(kh, diagK, gmax);
        linattn_kernel<<<N_NODES, 256, 0, stream>>>(qh, kh, vb, obuf);
        gemm(obuf, wo_p + (size_t)l * INNER * DIM, bo + l * DIM, nullptr, xf, nullptr, xf, DIM, INNER, 2);

        // --- feed-forward block ---
        ln_bf16_kernel<<<ROWS / 8, 256, 0, stream>>>(xf, ln_fg + l * DIM, ln_fb + l * DIM, hbuf);
        gemm(hbuf, w1_p + (size_t)l * DIM * FF, b1 + l * FF, nullptr, nullptr, ubuf, nullptr, FF, DIM, 1);
        gemm(ubuf, w2_p + (size_t)l * FF * DIM, b2 + l * DIM, nullptr, xf, nullptr, xf, DIM, FF, 2);
    }
}
